// DataSelectorCGCNN_30107720745196
// MI455X (gfx1250) — matrix operand order repaired and checked by reading, not compiled, not run on hardware
//
#include <hip/hip_runtime.h>

// ---------------------------------------------------------------------------
// Problem constants (fixed by setup_inputs in the reference)
// ---------------------------------------------------------------------------
#define FA     92
#define MM     12
#define FN     41
#define MAXN   10
#define SLOT   (FA + MM * FN + MM + 1)   // 597
#define DDIM   (MAXN * SLOT)             // 5970
#define KP     5984                      // DDIM padded up to multiple of 32
#define HDIM   2048

typedef __bf16 bf16_t;
typedef __attribute__((ext_vector_type(16))) __bf16 v16bf;
typedef __attribute__((ext_vector_type(8)))  __bf16 v8bf;
typedef __attribute__((ext_vector_type(8)))  float  v8f;

static __device__ __forceinline__ unsigned short f2bf_rne(float f) {
    unsigned int u = __float_as_uint(f);
    unsigned int lsb = (u >> 16) & 1u;
    u += 0x7fffu + lsb;                  // round-to-nearest-even
    return (unsigned short)(u >> 16);
}

// ---------------------------------------------------------------------------
// Kernel 0: zero the output accumulator, build fused head vector wc
// ---------------------------------------------------------------------------
__global__ void prep_kernel(const float* __restrict__ wp,
                            const float* __restrict__ wg,
                            const float* __restrict__ wphy,
                            const float* __restrict__ wgen,
                            float* __restrict__ wc,
                            float* __restrict__ out, int B) {
    int i = blockIdx.x * blockDim.x + threadIdx.x;
    if (i < HDIM) {
        wc[i] = wphy[0] * wp[i] + wgen[0] * wg[i];
    }
    int j = i - HDIM;
    if (j >= 0 && j < B) {
        out[j] = 0.0f;
    }
}

// ---------------------------------------------------------------------------
// Kernel 1: build padded/concatenated per-crystal feature row in bf16.
// One block per crystal; threads stride over the KP columns.
// ---------------------------------------------------------------------------
__global__ void build_feat_kernel(const float* __restrict__ atom_fea,
                                  const float* __restrict__ nbr_fea,
                                  const int*   __restrict__ nbr_idx,
                                  const int*   __restrict__ starts,
                                  const int*   __restrict__ lens,
                                  unsigned short* __restrict__ featbf,
                                  int N) {
    const int b     = blockIdx.x;
    const int start = starts[b];
    const int len   = lens[b];
    const int n_use = len < MAXN ? len : MAXN;
    unsigned short* row = featbf + (size_t)b * KP;

    for (int d = threadIdx.x; d < KP; d += blockDim.x) {
        float val = 0.0f;
        if (d < DDIM) {
            const int i = d / SLOT;          // atom slot within crystal
            const int j = d - i * SLOT;      // feature within slot
            const bool valid = (i < n_use);
            int pos = start + i;
            if (pos > N - 1) pos = N - 1;
            if (valid) {
                if (j < FA) {
                    val = atom_fea[(size_t)pos * FA + j];
                } else if (j < FA + MM * FN) {
                    const int q = j - FA;
                    const int m = q / FN;
                    const int f = q - m * FN;
                    val = nbr_fea[((size_t)pos * MM + m) * FN + f];
                } else if (j < FA + MM * FN + MM) {
                    const int m = j - (FA + MM * FN);
                    int nb = nbr_idx[(size_t)pos * MM + m] - start;
                    if (nb < 0) nb = 0;
                    if (nb >= n_use) nb = 0;
                    val = (float)nb * (1.0f / (float)MAXN);
                } else {
                    val = 1.0f;              // mask entry
                }
            }
        }
        row[d] = f2bf_rne(val);
    }
}

// ---------------------------------------------------------------------------
// Kernel 2: W1 [DDIM, HDIM] f32 row-major  ->  W1t [HDIM, KP] bf16
// (so each lane's B fragment is one contiguous, 32B-aligned 16-element load).
// LDS-tiled 32x32 transpose, coalesced on both sides; pads k in [DDIM,KP) = 0.
// ---------------------------------------------------------------------------
__global__ void transpose_w1_kernel(const float* __restrict__ W1,
                                    unsigned short* __restrict__ w1t) {
    __shared__ float tile[32][33];
    const int k0 = blockIdx.x * 32;          // gridDim.x = KP/32 = 187
    const int n0 = blockIdx.y * 32;          // gridDim.y = HDIM/32 = 64
#pragma unroll
    for (int i = 0; i < 4; ++i) {
        const int ky = threadIdx.y + i * 8;
        const int k  = k0 + ky;
        float v = 0.0f;
        if (k < DDIM) v = W1[(size_t)k * HDIM + n0 + threadIdx.x];
        tile[ky][threadIdx.x] = v;
    }
    __syncthreads();
#pragma unroll
    for (int i = 0; i < 4; ++i) {
        const int ny = threadIdx.y + i * 8;
        w1t[(size_t)(n0 + ny) * KP + (k0 + threadIdx.x)] =
            f2bf_rne(tile[threadIdx.x][ny]);
    }
}

// ---------------------------------------------------------------------------
// Kernel 3: register-blocked WMMA GEMM, fused bias + ReLU + head-dot.
//   h = relu(feat @ W1 + b1);  out[m] += sum_n h[m,n] * wc[n]
//
// Block: 256 threads = 8 wave32s arranged 2(M) x 4(N) -> 128x128 block tile.
// Each wave computes a 64x32 macro-tile = 4(M) x 2(N) grid of 16x16 WMMA
// tiles. Per K-step: 4 A fragments + 2 B fragments (12 b128 loads) feed
// 8 v_wmma_f32_16x16x32_bf16 (1.5 loads/WMMA). K loop unrolled x2 so the
// next step's loads issue underneath the current step's WMMAs (hides L2
// latency). Register footprint ~160 VGPRs under __launch_bounds__(256,1):
// spill-free.
// ---------------------------------------------------------------------------
__global__ void __launch_bounds__(256, 1)
gemm_wmma_kernel(const unsigned short* __restrict__ featbf,
                 const unsigned short* __restrict__ w1t,
                 const float* __restrict__ b1,
                 const float* __restrict__ wc,
                 float* __restrict__ out) {
    const int lane   = threadIdx.x & 31;
    const int wave   = threadIdx.x >> 5;
    const int wave_m = wave & 1;             // 2 waves along M
    const int wave_n = wave >> 1;            // 4 waves along N
    const int m_base = blockIdx.x * 128 + wave_m * 64;
    const int n_base = blockIdx.y * 128 + wave_n * 32;

    const int rowid = lane & 15;             // A row / B col / C col index
    const int hlf   = lane >> 4;             // half-wave selector

    // A fragment (16x32 bf16): lanes 0-15 carry K 0..15,
    // lanes 16-31 carry K 16..31, for row M = lane&15.
    const unsigned short* aBase =
        featbf + (size_t)(m_base + rowid) * KP + hlf * 16;
    // B fragment (32x16 bf16): lanes 0-15 col N = lane&15, K 0..15;
    // lanes 16-31 col N = lane&15, K 16..31. w1t is [H, KP]: contiguous
    // 32-byte aligned 16-element load per lane.
    const unsigned short* bBase =
        w1t + (size_t)(n_base + rowid) * KP + hlf * 16;

    v8f acc[4][2];
#pragma unroll
    for (int mi = 0; mi < 4; ++mi)
#pragma unroll
        for (int ni = 0; ni < 2; ++ni)
            acc[mi][ni] = (v8f){0.f, 0.f, 0.f, 0.f, 0.f, 0.f, 0.f, 0.f};

#pragma unroll 2
    for (int k = 0; k < KP; k += 32) {
        // prefetch fragments two K-steps ahead (global_prefetch_b8)
        __builtin_prefetch((const void*)(aBase + k + 64), 0, 1);
        __builtin_prefetch((const void*)(bBase + k + 64), 0, 1);

        v16bf a[4];
#pragma unroll
        for (int mi = 0; mi < 4; ++mi) {
            const unsigned short* p = aBase + (size_t)(mi * 16) * KP + k;
            v8bf lo = *(const v8bf*)(p);          // K k0    .. k0+7
            v8bf hi = *(const v8bf*)(p + 8);      // K k0+8  .. k0+15
            a[mi] = __builtin_shufflevector(lo, hi,
                                            0, 1, 2, 3, 4, 5, 6, 7,
                                            8, 9, 10, 11, 12, 13, 14, 15);
        }
        v16bf bm[2];
#pragma unroll
        for (int ni = 0; ni < 2; ++ni) {
            bm[ni] = *(const v16bf*)(bBase + (size_t)(ni * 16) * KP + k);
        }
#pragma unroll
        for (int mi = 0; mi < 4; ++mi)
#pragma unroll
            for (int ni = 0; ni < 2; ++ni)
                acc[mi][ni] = __builtin_amdgcn_wmma_f32_16x16x32_bf16(
                    false, a[mi], false, bm[ni],
                    (short)0, acc[mi][ni], false, false);
    }

    // Epilogue. Lane's column inside n-subtile ni is n_base + ni*16 + rowid;
    // its rows inside m-subtile mi are m_base + mi*16 + r + 8*hlf, r=0..7.
    float bias[2], wcv[2];
#pragma unroll
    for (int ni = 0; ni < 2; ++ni) {
        const int n = n_base + ni * 16 + rowid;
        bias[ni] = b1[n];
        wcv[ni]  = wc[n];
    }

#pragma unroll
    for (int mi = 0; mi < 4; ++mi) {
#pragma unroll
        for (int r = 0; r < 8; ++r) {
            float s = 0.0f;
#pragma unroll
            for (int ni = 0; ni < 2; ++ni) {
                float h = acc[mi][ni][r] + bias[ni];
                h = h > 0.0f ? h : 0.0f;
                s += h * wcv[ni];
            }
            // reduce across the 16 lanes of this half-wave (same output row)
#pragma unroll
            for (int off = 1; off < 16; off <<= 1) {
                s += __shfl_xor(s, off, 32);
            }
            if (rowid == 0) {
                atomicAdd(out + (m_base + mi * 16 + r + 8 * hlf), s);
            }
        }
    }
}

// ---------------------------------------------------------------------------
// Host-side launcher
// ---------------------------------------------------------------------------
extern "C" void kernel_launch(void* const* d_in, const int* in_sizes, int n_in,
                              void* d_out, int out_size, void* d_ws, size_t ws_size,
                              hipStream_t stream) {
    const float* atom_fea = (const float*)d_in[0];
    const float* nbr_fea  = (const float*)d_in[1];
    const float* W1       = (const float*)d_in[2];
    const float* b1       = (const float*)d_in[3];
    const float* wp       = (const float*)d_in[4];
    const float* wg       = (const float*)d_in[5];
    const float* wphy     = (const float*)d_in[6];
    const float* wgen     = (const float*)d_in[7];
    const int*   nbr_idx  = (const int*)d_in[8];
    const int*   starts   = (const int*)d_in[9];
    const int*   lens     = (const int*)d_in[10];

    const int B = in_sizes[9];               // 4096 crystals
    const int N = in_sizes[0] / FA;          // total atoms
    float* out = (float*)d_out;

    // workspace layout
    char* ws = (char*)d_ws;
    unsigned short* featbf = (unsigned short*)ws;                 // [B, KP] bf16
    size_t off = ((size_t)B * KP * sizeof(unsigned short) + 255) & ~(size_t)255;
    unsigned short* w1t = (unsigned short*)(ws + off);            // [H, KP] bf16
    off += ((size_t)HDIM * KP * sizeof(unsigned short) + 255) & ~(size_t)255;
    float* wc = (float*)(ws + off);                               // [H]

    // 0: zero output + fuse head vector
    {
        int total = HDIM + B;
        int blocks = (total + 255) / 256;
        prep_kernel<<<blocks, 256, 0, stream>>>(wp, wg, wphy, wgen, wc, out, B);
    }
    // 1: build features in bf16 (one block per crystal)
    build_feat_kernel<<<B, 256, 0, stream>>>(atom_fea, nbr_fea, nbr_idx,
                                             starts, lens, featbf, N);
    // 2: transpose + convert W1 to [H, KP] bf16
    {
        dim3 grid(KP / 32, HDIM / 32);
        dim3 block(32, 8);
        transpose_w1_kernel<<<grid, block, 0, stream>>>(W1, w1t);
    }
    // 3: register-blocked fused WMMA GEMM (128x128 block tile)
    {
        dim3 grid(B / 128, HDIM / 128);
        gemm_wmma_kernel<<<grid, 256, 0, stream>>>(featbf, w1t, b1, wc, out);
    }
}
